// UltimateDualPathTNN_13280038879581
// MI455X (gfx1250) — compile-verified, matrix operand order checked by reading, not run on hardware
//
#include <hip/hip_runtime.h>
#include <cstddef>

// ---------------- problem constants (match reference) ----------------
#define LNUM  4
#define NBLK  8
#define BATCH 16
#define SEQ   512
#define DIM   512
#define TOK   (BATCH * SEQ)      // 8192 tokens

// ---------------- WMMA types (CDNA5 gfx1250, wave32) ----------------
typedef __attribute__((ext_vector_type(16))) __bf16 v16bf;
typedef __attribute__((ext_vector_type(8)))  float  v8f;

union AFrag { v16bf v; uint4 q[2]; };

// async global->LDS path (probe with __has_builtin; sync fallback otherwise)
#if defined(__HIP_DEVICE_COMPILE__) && __has_builtin(__builtin_amdgcn_global_load_async_to_lds_b128)
#define USE_ASYNC 1
typedef int v4i_ __attribute__((__vector_size__(4 * sizeof(int))));
typedef __attribute__((address_space(1))) v4i_ g_v4i;   // global (prints as __device__)
typedef __attribute__((address_space(3))) v4i_ l_v4i;   // LDS    (prints as __shared__)
#define GPTR(p) ((g_v4i*)(p))
#define LPTR(p) ((l_v4i*)(p))
#if __has_builtin(__builtin_amdgcn_s_wait_asynccnt)
#define WAIT_ASYNC(n) __builtin_amdgcn_s_wait_asynccnt(n)
#else
#define WAIT_ASYNC(n) asm volatile("s_wait_asynccnt %0" ::"n"(n) : "memory")
#endif
#else
#define USE_ASYNC 0
#endif

__device__ __forceinline__ unsigned short f2bf(float f) {
  union { float f; unsigned int u; } c; c.f = f;
  unsigned int u = c.u;
  unsigned int r = u + 0x7FFFu + ((u >> 16) & 1u);   // round-to-nearest-even
  return (unsigned short)(r >> 16);
}
__device__ __forceinline__ uint2 pack4(float4 f) {
  uint2 p;
  p.x = (unsigned)f2bf(f.x) | ((unsigned)f2bf(f.y) << 16);
  p.y = (unsigned)f2bf(f.z) | ((unsigned)f2bf(f.w) << 16);
  return p;
}
__device__ __forceinline__ float sigmoidf_(float x) { return 1.0f / (1.0f + expf(-x)); }

// ---------------- per-batch token mean:  x[B,S,D] -> mean[B,D] ----------------
__global__ void mean_kernel(const float* __restrict__ x, float* __restrict__ mean,
                            int Sdim, int Dd) {
  int b = blockIdx.x;
  int d = threadIdx.x;                         // blockDim = DIM (512)
  const float* xp = x + (size_t)b * Sdim * Dd + d;
  float s = 0.0f;
  for (int i = 0; i < Sdim; ++i) s += xp[(size_t)i * Dd];
  mean[b * Dd + d] = s / (float)Sdim;
}

// ---------------- router: scores on sample 0, top-2 (JAX tie rules) ----------------
__global__ __launch_bounds__(256)
void router_kernel(const float* __restrict__ mean0, const float* __restrict__ wsel,
                   const float* __restrict__ bsel, int* __restrict__ idxOut,
                   int Dd, int NBd) {
  int wave = threadIdx.x >> 5;                 // wave32: 8 waves = NB blocks
  int lane = threadIdx.x & 31;
  float s = 0.0f;
  for (int d = lane; d < Dd; d += 32) s += mean0[d] * wsel[d * NBd + wave];
  for (int off = 16; off > 0; off >>= 1) s += __shfl_xor(s, off, 32);
  __shared__ float sc[NBLK];
  if (lane == 0 && wave < NBd) sc[wave] = s + bsel[wave];
  __syncthreads();
  if (threadIdx.x == 0) {
    float adj[NBLK];
    for (int n = 0; n < NBd; ++n) adj[n] = 0.7f * sigmoidf_(sc[n]) + 0.15f;
    int b0 = 0;
    for (int n = 1; n < NBd; ++n) if (adj[n] > adj[b0]) b0 = n;
    int b1 = -1;
    for (int n = 0; n < NBd; ++n) {
      if (n == b0) continue;
      if (b1 < 0 || adj[n] > adj[b1]) b1 = n;
    }
    idxOut[0] = b0; idxOut[1] = b1;
  }
}

// ---------------- LayerNorm of selected block -> bf16 activations ----------------
__global__ __launch_bounds__(256)
void ln_kernel(const float* __restrict__ x, const float* __restrict__ gbase,
               const float* __restrict__ bbase, const int* __restrict__ idx, int jsel,
               unsigned short* __restrict__ out, int Dd) {
  int row = blockIdx.x;
  int t = threadIdx.x;                          // 256 threads, 2 elems each (D=512)
  const float* xr = x + (size_t)row * Dd;
  float v0 = xr[t], v1 = xr[t + 256];
  __shared__ float red[256], red2[256];
  red[t]  = v0 + v1;
  red2[t] = v0 * v0 + v1 * v1;
  __syncthreads();
  for (int off = 128; off > 0; off >>= 1) {
    if (t < off) { red[t] += red[t + off]; red2[t] += red2[t + off]; }
    __syncthreads();
  }
  __shared__ float smu, srs;
  if (t == 0) {
    float mu  = red[0] / (float)Dd;
    float var = red2[0] / (float)Dd - mu * mu;
    smu = mu; srs = rsqrtf(var + 1e-5f);
  }
  __syncthreads();
  float mu = smu, rs = srs;
  size_t go = (size_t)idx[jsel] * Dd;
  const float* g  = gbase + go;
  const float* be = bbase + go;
  out[(size_t)row * Dd + t]       = f2bf((v0 - mu) * rs * g[t]       + be[t]);
  out[(size_t)row * Dd + t + 256] = f2bf((v1 - mu) * rs * g[t + 256] + be[t + 256]);
}

// -------- weight convert+transpose of the routed block: W[K,N] f32 -> WT[N,K] bf16 --------
__global__ __launch_bounds__(256)
void wconv_kernel(const float* __restrict__ W, const int* __restrict__ idx, int jsel,
                  size_t blkStride, int N, int K, unsigned short* __restrict__ WT) {
  __shared__ unsigned short tile[32][33];
  const float* Wp = W + (idx ? (size_t)idx[jsel] * blkStride : 0);
  int k0 = blockIdx.x * 32, n0 = blockIdx.y * 32;
  int tid = threadIdx.x;
  int kr = tid >> 3, c4 = (tid & 7) << 2;
  float4 f = *reinterpret_cast<const float4*>(Wp + (size_t)(k0 + kr) * N + n0 + c4);
  tile[kr][c4 + 0] = f2bf(f.x);
  tile[kr][c4 + 1] = f2bf(f.y);
  tile[kr][c4 + 2] = f2bf(f.z);
  tile[kr][c4 + 3] = f2bf(f.w);
  __syncthreads();
  int nr = tid >> 3;                          // same decomposition, transposed roles
  uint2 p;
  p.x = (unsigned)tile[c4 + 0][nr] | ((unsigned)tile[c4 + 1][nr] << 16);
  p.y = (unsigned)tile[c4 + 2][nr] | ((unsigned)tile[c4 + 3][nr] << 16);
  *reinterpret_cast<uint2*>(WT + (size_t)(n0 + nr) * K + k0 + c4) = p;
}

// -------- concat-convert: [c | b] fp32 rows -> bf16 [T, 2D] for the cross GEMM --------
__global__ __launch_bounds__(256)
void cat_bf16_kernel(const float* __restrict__ a, const float* __restrict__ b,
                     unsigned short* __restrict__ out, int Dd) {
  int idx = blockIdx.x * 256 + threadIdx.x;    // over TOK * Dd / 4 chunks
  int row = idx >> 7;                          // Dd/4 = 128 chunks per row
  int c4  = (idx & 127) << 2;
  float4 f = *reinterpret_cast<const float4*>(a + (size_t)row * Dd + c4);
  float4 g = *reinterpret_cast<const float4*>(b + (size_t)row * Dd + c4);
  unsigned short* o = out + (size_t)row * (2 * Dd);
  *reinterpret_cast<uint2*>(o + c4)      = pack4(f);
  *reinterpret_cast<uint2*>(o + Dd + c4) = pack4(g);
}

// ---------------- WMMA GEMM: C = act(A @ Wt^T + bias) with fused epilogues ----------------
// A: bf16 [M,K] row-major.  Bt: bf16 [N,K] row-major (pre-transposed weights).
// Tiles 128x128x32, 256 threads = 8 waves (2M x 4N), each wave 4x2 WMMA 16x16 tiles.
// EPI: 0 = GELU(erf)->bf16; 1 = tanh->bf16; 2 = torsion-gate+fusion+residual->f32; 3 = bias->f32
#define BMT 128
#define BNT 128
#define BKT 32
#define LDSS 40                       // padded LDS row stride (elements; 80B, 16B-aligned)
#define NBUF (USE_ASYNC ? 2 : 1)

template<int EPI>
__global__ __launch_bounds__(256)
void gemm_wmma_kernel(const unsigned short* __restrict__ Abf,
                      const unsigned short* __restrict__ Bt,
                      const float* __restrict__ bias,
                      const int* __restrict__ idx, int jsel,
                      int N, int KDIM,
                      unsigned short* __restrict__ outH,
                      float* __restrict__ outF, const float* __restrict__ xin,
                      const float* __restrict__ gate, const float* __restrict__ torsion,
                      const float* __restrict__ sigMean, int Sdim) {
  __shared__ __align__(16) unsigned short sA[NBUF][BMT * LDSS];
  __shared__ __align__(16) unsigned short sB[NBUF][BNT * LDSS];

  const int tid   = threadIdx.x;
  const int mBase = blockIdx.x * BMT;
  const int nBase = blockIdx.y * BNT;

  size_t blkOff = 0;
  if (idx) blkOff = (size_t)idx[jsel];                 // device-side block routing
  const float* bp = bias + blkOff * (size_t)N;
  const float* gp = (EPI == 2) ? (gate + blkOff * (size_t)N) : nullptr;

  const unsigned lane  = tid & 31u;
  const unsigned wave  = (unsigned)tid >> 5;
  const unsigned waveM = wave >> 2, waveN = wave & 3u;
  const unsigned laneM = lane & 15u, hi = lane >> 4;

  // this thread's two 16B chunks per tile (512 chunks of 8 bf16 per 128x32 tile)
  const int c0 = tid, c1 = tid + 256;
  const int ar0 = c0 >> 2, as0 = (c0 & 3) << 3;
  const int ar1 = c1 >> 2, as1 = (c1 & 3) << 3;
  const unsigned short* Arow = Abf + (size_t)mBase * KDIM;
  const unsigned short* Brow = Bt  + (size_t)nBase * KDIM;

  v8f acc[4][2];
#pragma unroll
  for (int m = 0; m < 4; ++m)
#pragma unroll
    for (int n = 0; n < 2; ++n)
#pragma unroll
      for (int e = 0; e < 8; ++e) acc[m][n][e] = 0.0f;

  const int nk = KDIM / BKT;

#if USE_ASYNC
  auto issue = [&](int buf, int k0) {
    __builtin_amdgcn_global_load_async_to_lds_b128(
        GPTR(Arow + (size_t)ar0 * KDIM + k0 + as0),
        LPTR(&sA[buf][ar0 * LDSS + as0]), 0, 0);
    __builtin_amdgcn_global_load_async_to_lds_b128(
        GPTR(Arow + (size_t)ar1 * KDIM + k0 + as1),
        LPTR(&sA[buf][ar1 * LDSS + as1]), 0, 0);
    __builtin_amdgcn_global_load_async_to_lds_b128(
        GPTR(Brow + (size_t)ar0 * KDIM + k0 + as0),
        LPTR(&sB[buf][ar0 * LDSS + as0]), 0, 0);
    __builtin_amdgcn_global_load_async_to_lds_b128(
        GPTR(Brow + (size_t)ar1 * KDIM + k0 + as1),
        LPTR(&sB[buf][ar1 * LDSS + as1]), 0, 0);
  };
  issue(0, 0);
#endif

  for (int kt = 0; kt < nk; ++kt) {
    const int k0 = kt * BKT;
#if USE_ASYNC
    const int buf = kt & 1;
    if (kt + 1 < nk) {
      issue(buf ^ 1, k0 + BKT);               // prefetch next stage into other buffer
      WAIT_ASYNC(4);                          // first 4 (current stage) complete in order
    } else {
      WAIT_ASYNC(0);
    }
    __syncthreads();
#else
    const int buf = 0;
    uint4 ra0 = *reinterpret_cast<const uint4*>(Arow + (size_t)ar0 * KDIM + k0 + as0);
    uint4 ra1 = *reinterpret_cast<const uint4*>(Arow + (size_t)ar1 * KDIM + k0 + as1);
    uint4 rb0 = *reinterpret_cast<const uint4*>(Brow + (size_t)ar0 * KDIM + k0 + as0);
    uint4 rb1 = *reinterpret_cast<const uint4*>(Brow + (size_t)ar1 * KDIM + k0 + as1);
    if (kt + 1 < nk)   // global_prefetch_b8 of next tiles
      __builtin_prefetch(Brow + (size_t)ar0 * KDIM + k0 + BKT + as0, 0, 0);
    __syncthreads();   // previous stage fully consumed before overwriting
    *reinterpret_cast<uint4*>(&sA[0][ar0 * LDSS + as0]) = ra0;
    *reinterpret_cast<uint4*>(&sA[0][ar1 * LDSS + as1]) = ra1;
    *reinterpret_cast<uint4*>(&sB[0][ar0 * LDSS + as0]) = rb0;
    *reinterpret_cast<uint4*>(&sB[0][ar1 * LDSS + as1]) = rb1;
    __syncthreads();
#endif

    // fragments per ISA 16-bit A 16x32 layout: lo lanes K0-7/K16-23, hi lanes K8-15/K24-31
    const unsigned short* pA = sA[buf];
    const unsigned short* pB = sB[buf];
    AFrag a[4], b[2];
#pragma unroll
    for (int m = 0; m < 4; ++m) {
      int base = (int)((waveM * 64u + m * 16u + laneM) * LDSS + hi * 8u);
      a[m].q[0] = *reinterpret_cast<const uint4*>(&pA[base]);
      a[m].q[1] = *reinterpret_cast<const uint4*>(&pA[base + 16]);
    }
#pragma unroll
    for (int n = 0; n < 2; ++n) {
      int base = (int)((waveN * 32u + n * 16u + laneM) * LDSS + hi * 8u);
      b[n].q[0] = *reinterpret_cast<const uint4*>(&pB[base]);
      b[n].q[1] = *reinterpret_cast<const uint4*>(&pB[base + 16]);
    }
#pragma unroll
    for (int m = 0; m < 4; ++m)
#pragma unroll
      for (int n = 0; n < 2; ++n)
        acc[m][n] = __builtin_amdgcn_wmma_f32_16x16x32_bf16(
            false, a[m].v, false, b[n].v, (short)0, acc[m][n], false, false);
    __syncthreads();
  }

  // ---- epilogue: C/D layout -> row = r + hi*8, col = laneM within each 16x16 tile
#pragma unroll
  for (int m = 0; m < 4; ++m) {
    int row0  = mBase + (int)(waveM * 64u + m * 16u + hi * 8u);
    int batch = row0 / Sdim;                           // 8-row span never straddles a batch
#pragma unroll
    for (int n = 0; n < 2; ++n) {
      int col  = nBase + (int)(waveN * 32u + n * 16u + laneM);
      float bv = bp[col];
      float tg = 0.0f, tors = 0.0f, sgl = 0.0f;
      if (EPI == 2) {
        tg   = sigmoidf_(gp[col]);
        tors = torsion[col];
        sgl  = sigMean[batch * N + col] * 0.3f * tg;   // fusion signal
      }
#pragma unroll
      for (int vr = 0; vr < 8; ++vr) {
        int row  = row0 + vr;
        float v  = acc[m][n][vr] + bv;
        size_t o = (size_t)row * N + col;
        if (EPI == 0) {
          outH[o] = f2bf(0.5f * v * (1.0f + erff(v * 0.70710678118654752f)));
        } else if (EPI == 1) {
          outH[o] = f2bf(tanhf(v));
        } else if (EPI == 2) {
          v = v * (1.0f + tg * tors) + sgl;
          outF[o] = xin[o] + 0.5f * v;                 // residual, in place
        } else {
          outF[o] = v;
        }
      }
    }
  }
}

// ---------------- host orchestration ----------------
extern "C" void kernel_launch(void* const* d_in, const int* in_sizes, int n_in,
                              void* d_out, int out_size, void* d_ws, size_t ws_size,
                              hipStream_t stream) {
  (void)in_sizes; (void)n_in; (void)out_size; (void)ws_size;
  const float* cortical  = (const float*)d_in[0];
  const float* brainstem = (const float*)d_in[1];
  const float* torsion   = (const float*)d_in[2];
  const float* w_sel_c   = (const float*)d_in[3];
  const float* b_sel_c   = (const float*)d_in[4];
  const float* w_sel_b   = (const float*)d_in[5];
  const float* b_sel_b   = (const float*)d_in[6];
  const float* c_ln_g    = (const float*)d_in[7];
  const float* c_ln_b    = (const float*)d_in[8];
  const float* c_w1      = (const float*)d_in[9];
  const float* c_b1      = (const float*)d_in[10];
  const float* c_w2      = (const float*)d_in[11];
  const float* c_b2      = (const float*)d_in[12];
  const float* c_gate    = (const float*)d_in[13];
  const float* b_ln_g    = (const float*)d_in[14];
  const float* b_ln_b    = (const float*)d_in[15];
  const float* b_w1      = (const float*)d_in[16];
  const float* b_b1      = (const float*)d_in[17];
  const float* b_w2      = (const float*)d_in[18];
  const float* b_b2      = (const float*)d_in[19];
  const float* b_gate    = (const float*)d_in[20];
  const float* w_cross   = (const float*)d_in[21];
  const float* b_cross   = (const float*)d_in[22];

  float* outC     = (float*)d_out;                  // c lives here in place
  float* outB     = outC + (size_t)TOK * DIM;       // b lives here in place
  float* outFused = outB + (size_t)TOK * DIM;

  unsigned char* w = (unsigned char*)d_ws;
  unsigned short* xn  = (unsigned short*)w; w += (size_t)TOK * DIM * 2;       // LN out (bf16)
  unsigned short* h1  = (unsigned short*)w; w += (size_t)TOK * 2 * DIM * 2;   // hidden (bf16)
  unsigned short* wt1 = (unsigned short*)w; w += (size_t)2 * DIM * DIM * 2;   // W1^T bf16
  unsigned short* wt2 = (unsigned short*)w; w += (size_t)2 * DIM * DIM * 2;   // W2^T bf16
  float* meanC = (float*)w; w += (size_t)BATCH * DIM * 4;
  float* meanB = (float*)w; w += (size_t)BATCH * DIM * 4;
  int* idxC = (int*)w; w += 256;
  int* idxB = (int*)w; w += 256;

  const size_t tensorBytes = (size_t)TOK * DIM * sizeof(float);
  (void)hipMemcpyAsync(outC, cortical,  tensorBytes, hipMemcpyDeviceToDevice, stream);
  (void)hipMemcpyAsync(outB, brainstem, tensorBytes, hipMemcpyDeviceToDevice, stream);

  for (int l = 0; l < LNUM; ++l) {
    // means of layer inputs (router input + fusion signals; both pre-update)
    mean_kernel<<<BATCH, DIM, 0, stream>>>(outC, meanC, SEQ, DIM);
    mean_kernel<<<BATCH, DIM, 0, stream>>>(outB, meanB, SEQ, DIM);
    router_kernel<<<1, 256, 0, stream>>>(meanC, w_sel_c + (size_t)l * DIM * NBLK,
                                         b_sel_c + (size_t)l * NBLK, idxC, DIM, NBLK);
    router_kernel<<<1, 256, 0, stream>>>(meanB, w_sel_b + (size_t)l * DIM * NBLK,
                                         b_sel_b + (size_t)l * NBLK, idxB, DIM, NBLK);

    // cortical pathway: K=2 routed blocks, sequential
    for (int j = 0; j < 2; ++j) {
      wconv_kernel<<<dim3(DIM / 32, 2 * DIM / 32), 256, 0, stream>>>(
          c_w1 + (size_t)l * NBLK * DIM * 2 * DIM, idxC, j,
          (size_t)DIM * 2 * DIM, 2 * DIM, DIM, wt1);
      ln_kernel<<<TOK, 256, 0, stream>>>(outC, c_ln_g + (size_t)l * NBLK * DIM,
                                         c_ln_b + (size_t)l * NBLK * DIM, idxC, j, xn, DIM);
      gemm_wmma_kernel<0><<<dim3(TOK / BMT, (2 * DIM) / BNT), 256, 0, stream>>>(
          xn, wt1, c_b1 + (size_t)l * NBLK * 2 * DIM, idxC, j, 2 * DIM, DIM,
          h1, nullptr, nullptr, nullptr, nullptr, nullptr, SEQ);
      wconv_kernel<<<dim3(2 * DIM / 32, DIM / 32), 256, 0, stream>>>(
          c_w2 + (size_t)l * NBLK * 2 * DIM * DIM, idxC, j,
          (size_t)2 * DIM * DIM, DIM, 2 * DIM, wt2);
      gemm_wmma_kernel<2><<<dim3(TOK / BMT, DIM / BNT), 256, 0, stream>>>(
          h1, wt2, c_b2 + (size_t)l * NBLK * DIM, idxC, j, DIM, 2 * DIM,
          nullptr, outC, outC, c_gate + (size_t)l * NBLK * DIM, torsion, meanB, SEQ);
    }
    // brainstem pathway (uses pre-update c via meanC)
    for (int j = 0; j < 2; ++j) {
      wconv_kernel<<<dim3(DIM / 32, DIM / 32), 256, 0, stream>>>(
          b_w1 + (size_t)l * NBLK * DIM * DIM, idxB, j,
          (size_t)DIM * DIM, DIM, DIM, wt1);
      ln_kernel<<<TOK, 256, 0, stream>>>(outB, b_ln_g + (size_t)l * NBLK * DIM,
                                         b_ln_b + (size_t)l * NBLK * DIM, idxB, j, xn, DIM);
      gemm_wmma_kernel<1><<<dim3(TOK / BMT, DIM / BNT), 256, 0, stream>>>(
          xn, wt1, b_b1 + (size_t)l * NBLK * DIM, idxB, j, DIM, DIM,
          h1, nullptr, nullptr, nullptr, nullptr, nullptr, SEQ);
      wconv_kernel<<<dim3(DIM / 32, DIM / 32), 256, 0, stream>>>(
          b_w2 + (size_t)l * NBLK * DIM * DIM, idxB, j,
          (size_t)DIM * DIM, DIM, DIM, wt2);
      gemm_wmma_kernel<2><<<dim3(TOK / BMT, DIM / BNT), 256, 0, stream>>>(
          h1, wt2, b_b2 + (size_t)l * NBLK * DIM, idxB, j, DIM, DIM,
          nullptr, outB, outB, b_gate + (size_t)l * NBLK * DIM, torsion, meanC, SEQ);
    }
  }
  // only the last layer's fused output is returned: fused = [c|b] @ w_cross + b_cross
  cat_bf16_kernel<<<(TOK * DIM / 4) / 256, 256, 0, stream>>>(outC, outB, h1, DIM);
  wconv_kernel<<<dim3(2 * DIM / 32, DIM / 32), 256, 0, stream>>>(
      w_cross + (size_t)(LNUM - 1) * 2 * DIM * DIM, nullptr, 0,
      0, DIM, 2 * DIM, wt1);
  gemm_wmma_kernel<3><<<dim3(TOK / BMT, DIM / BNT), 256, 0, stream>>>(
      h1, wt1, b_cross + (size_t)(LNUM - 1) * DIM, nullptr, 0, DIM, 2 * DIM,
      nullptr, outFused, nullptr, nullptr, nullptr, nullptr, SEQ);
}
